// T5Attention_4655744549114
// MI455X (gfx1250) — matrix operand order repaired and checked by reading, not compiled, not run on hardware
//
#include <hip/hip_runtime.h>

// ---------------- problem constants ----------------
#define EMBED 1024
#define HEADS 16
#define HEAD_DIM 64
#define SEQ 2048
#define BATCH 4
#define MROWS (BATCH * SEQ) // 8192 GEMM rows

typedef __bf16  v16bf __attribute__((ext_vector_type(16)));
typedef __bf16  v8bf  __attribute__((ext_vector_type(8)));
typedef float   v8f   __attribute__((ext_vector_type(8)));

__device__ __forceinline__ v8f wmma_bf16(v16bf a, v16bf b, v8f c) {
  // D = A(16x32 bf16) * B(32x16 bf16) + C(16x16 f32)
  return __builtin_amdgcn_wmma_f32_16x16x32_bf16(false, a, false, b, (short)0, c, false, false);
}

__device__ __forceinline__ v16bf cat8(v8bf lo, v8bf hi) {
  v16bf r;
#pragma unroll
  for (int i = 0; i < 8; ++i) { r[i] = lo[i]; r[8 + i] = hi[i]; }
  return r;
}

// A-matrix fragment (16x32, rows = lane&15): lanes 0-15 hold K [0..15],
// lanes 16-31 hold K [16..31]  (contiguous order, matches B loads)
__device__ __forceinline__ v16bf load_a_frag(const __bf16* row_ptr, int hf) {
  const __bf16* p = row_ptr + 16 * hf;
  return *(const v16bf*)p;
}

// ---------------- f32 -> bf16 convert ----------------
__global__ void cvt_kernel(const float* __restrict__ src, __bf16* __restrict__ dst, int n) {
  int i = blockIdx.x * blockDim.x + threadIdx.x;
  if (i < n) dst[i] = (__bf16)src[i];
}

// ---------------- T5 relative bias table: bias_full[h][delta], delta=q-k clamped >=0 ----
__global__ void bias_kernel(const float* __restrict__ rel_bias, float* __restrict__ bias_full) {
  int i = blockIdx.x * blockDim.x + threadIdx.x; // i = h*SEQ + n
  if (i >= HEADS * SEQ) return;
  int h = i / SEQ, n = i % SEQ;
  int bucket;
  if (n < 16) {
    bucket = n;
  } else {
    int v = 16 + (int)(__logf((float)n * (1.0f / 16.0f)) * (16.0f / __logf(8.0f)));
    bucket = v < 31 ? v : 31;
  }
  bias_full[i] = rel_bias[bucket * HEADS + h];
}

// ================================================================
// LDS-staged double-buffered GEMM core: 128x128 block tile, K-step 32,
// 8 waves (4M x 2N), wave tile 32x64 (eight 16x16x32 WMMAs per K-step).
// Both X rows and W rows are K-contiguous (y = x @ W^T), matching the
// WMMA A/B per-lane fragment layouts directly.
// ================================================================
typedef __bf16 tile_t[2][128][32];

__device__ __forceinline__ void gemm_core(const __bf16* __restrict__ X,
                                          const __bf16* __restrict__ W,
                                          tile_t& lA, tile_t& lB,
                                          int mBlk, int nBlk,
                                          int waveM, int waveN, int l16, int hf,
                                          v8f acc[2][4]) {
  const int srow = threadIdx.x >> 1;       // 0..127 staging row
  const int scol = (threadIdx.x & 1) * 16; // 0 / 16 staging chunk

  // prologue: stage tile kk=0 into buffer 0
  *(v16bf*)&lA[0][srow][scol] = *(const v16bf*)(X + (size_t)(mBlk + srow) * EMBED + scol);
  *(v16bf*)&lB[0][srow][scol] = *(const v16bf*)(W + (size_t)(nBlk + srow) * EMBED + scol);
  __syncthreads();

  for (int kk = 0; kk < EMBED; kk += 32) {
    const int buf = (kk >> 5) & 1;
    if (kk + 32 < EMBED) {
      *(v16bf*)&lA[buf ^ 1][srow][scol] =
          *(const v16bf*)(X + (size_t)(mBlk + srow) * EMBED + kk + 32 + scol);
      *(v16bf*)&lB[buf ^ 1][srow][scol] =
          *(const v16bf*)(W + (size_t)(nBlk + srow) * EMBED + kk + 32 + scol);
      if (kk + 64 < EMBED) // WGP-scope prefetch of the tile after next
        __builtin_prefetch(X + (size_t)(mBlk + srow) * EMBED + kk + 64, 0, 3);
    }
    v16bf a[2], bf[4];
#pragma unroll
    for (int mi = 0; mi < 2; ++mi) {
      const __bf16* pr = &lA[buf][waveM + mi * 16 + l16][0];
      a[mi] = *(const v16bf*)(pr + 16 * hf);
    }
#pragma unroll
    for (int ni = 0; ni < 4; ++ni)
      bf[ni] = *(const v16bf*)&lB[buf][waveN + ni * 16 + l16][16 * hf];
#pragma unroll
    for (int mi = 0; mi < 2; ++mi)
#pragma unroll
      for (int ni = 0; ni < 4; ++ni)
        acc[mi][ni] = wmma_bf16(a[mi], bf[ni], acc[mi][ni]);
    __syncthreads();
  }
}

// ---------------- QKV projection GEMM: y = x @ W^T + b ----------------
// mode 0: Q -> [b,h,l,d] scaled 1/8 ; mode 1: K -> [b,h,l,d] ; mode 2: V -> [b,h,d,l]
__global__ __launch_bounds__(256) void proj_kernel(const __bf16* __restrict__ X,
                                                   const __bf16* __restrict__ W,
                                                   const float* __restrict__ bias,
                                                   __bf16* __restrict__ dst, int mode) {
  __shared__ __align__(32) tile_t lA;
  __shared__ __align__(32) tile_t lB;
  const int lane = threadIdx.x & 31;
  const int wave = threadIdx.x >> 5;
  const int waveM = (wave & 3) * 32;
  const int waveN = (wave >> 2) * 64;
  const int mBlk = blockIdx.y * 128;
  const int nBlk = blockIdx.x * 128;
  const int hf = lane >> 4, l16 = lane & 15;

  v8f acc[2][4] = {};
  gemm_core(X, W, lA, lB, mBlk, nBlk, waveM, waveN, l16, hf, acc);

  const float scale = (mode == 0) ? 0.125f : 1.0f; // fold 1/sqrt(64) into Q
#pragma unroll
  for (int mi = 0; mi < 2; ++mi)
#pragma unroll
    for (int ni = 0; ni < 4; ++ni)
#pragma unroll
      for (int r = 0; r < 8; ++r) {
        int m = mBlk + waveM + mi * 16 + r + 8 * hf; // global row = b*SEQ + l
        int n = nBlk + waveN + ni * 16 + l16;        // channel = h*64 + d
        float v = (acc[mi][ni][r] + bias[n]) * scale;
        int bb = m >> 11;
        int l = m & (SEQ - 1);
        int h = n >> 6;
        int d = n & 63;
        if (mode == 2)
          dst[(((size_t)(bb * HEADS + h) * HEAD_DIM + d) * SEQ) + l] = (__bf16)v; // V^T
        else
          dst[(((size_t)(bb * HEADS + h) * SEQ + l) * HEAD_DIM) + d] = (__bf16)v;
      }
}

// ---------------- output projection: out = ctx @ Wo^T + bo (f32 out) ----------------
__global__ __launch_bounds__(256) void outproj_kernel(const __bf16* __restrict__ X,
                                                      const __bf16* __restrict__ W,
                                                      const float* __restrict__ bias,
                                                      float* __restrict__ out) {
  __shared__ __align__(32) tile_t lA;
  __shared__ __align__(32) tile_t lB;
  const int lane = threadIdx.x & 31;
  const int wave = threadIdx.x >> 5;
  const int waveM = (wave & 3) * 32;
  const int waveN = (wave >> 2) * 64;
  const int mBlk = blockIdx.y * 128;
  const int nBlk = blockIdx.x * 128;
  const int hf = lane >> 4, l16 = lane & 15;

  v8f acc[2][4] = {};
  gemm_core(X, W, lA, lB, mBlk, nBlk, waveM, waveN, l16, hf, acc);

#pragma unroll
  for (int mi = 0; mi < 2; ++mi)
#pragma unroll
    for (int ni = 0; ni < 4; ++ni)
#pragma unroll
      for (int r = 0; r < 8; ++r) {
        int m = mBlk + waveM + mi * 16 + r + 8 * hf;
        int n = nBlk + waveN + ni * 16 + l16;
        out[(size_t)m * EMBED + n] = acc[mi][ni][r] + bias[n];
      }
}

// ---------------- flash attention: softmax(Q K^T + bias) V ----------------
// Q,K: [b,h,l,d] bf16 (Q pre-scaled), Vt: [b,h,d,l] bf16, ctx out: [b,l,h*64+d] bf16
__global__ __launch_bounds__(256) void attn_kernel(const __bf16* __restrict__ Q,
                                                   const __bf16* __restrict__ K,
                                                   const __bf16* __restrict__ Vt,
                                                   const float* __restrict__ bias_full,
                                                   __bf16* __restrict__ ctx) {
  __shared__ float sbias[SEQ];
  __shared__ __align__(32) __bf16 pbuf[8][16][32];

  const int b = blockIdx.z, h = blockIdx.y;
  const int wave = threadIdx.x >> 5, lane = threadIdx.x & 31;
  const int l16 = lane & 15, hf = lane >> 4;

  for (int i = threadIdx.x; i < SEQ; i += 256) sbias[i] = bias_full[h * SEQ + i];
  __syncthreads();

  const __bf16* Qb = Q + (size_t)(b * HEADS + h) * SEQ * HEAD_DIM;
  const __bf16* Kb = K + (size_t)(b * HEADS + h) * SEQ * HEAD_DIM;
  const __bf16* Vb = Vt + (size_t)(b * HEADS + h) * HEAD_DIM * SEQ;
  const int qBase = blockIdx.x * 128 + wave * 16;

  // Q fragments: 16 rows x K=64 -> two 16x32 A frags
  v16bf qf0 = load_a_frag(Qb + (size_t)(qBase + l16) * HEAD_DIM + 0, hf);
  v16bf qf1 = load_a_frag(Qb + (size_t)(qBase + l16) * HEAD_DIM + 32, hf);

  v8f o[4] = {};
  float rmax[8], rsum[8];
#pragma unroll
  for (int r = 0; r < 8; ++r) { rmax[r] = -1e30f; rsum[r] = 0.0f; }

  for (int kt = 0; kt < SEQ; kt += 32) {
    // ---- S = Q K^T tile (16 x 32) : B col n = K row (kt+n), contiguous d per lane
    const __bf16* kp0 = Kb + (size_t)(kt + l16) * HEAD_DIM + 16 * hf;
    v16bf kb00 = *(const v16bf*)kp0;
    v16bf kb01 = *(const v16bf*)(kp0 + 32);
    const __bf16* kp1 = kp0 + 16 * HEAD_DIM;
    v16bf kb10 = *(const v16bf*)kp1;
    v16bf kb11 = *(const v16bf*)(kp1 + 32);
    v8f s0 = {}, s1 = {};
    s0 = wmma_bf16(qf0, kb00, s0);
    s0 = wmma_bf16(qf1, kb01, s0);
    s1 = wmma_bf16(qf0, kb10, s1);
    s1 = wmma_bf16(qf1, kb11, s1);

    if (kt + 32 < SEQ) { // WGP-scope: all 8 waves of this block reuse these lines
      __builtin_prefetch(Kb + (size_t)(kt + 32 + l16) * HEAD_DIM, 0, 3);
      __builtin_prefetch(Vb + (size_t)l16 * SEQ + kt + 32, 0, 3);
    }

    // ---- online softmax (row r+8*hf lives in this lane's VGPR slot r)
#pragma unroll
    for (int r = 0; r < 8; ++r) {
      int qrow = qBase + r + 8 * hf;
      int d0 = qrow - (kt + l16);      if (d0 < 0) d0 = 0;
      int d1 = qrow - (kt + 16 + l16); if (d1 < 0) d1 = 0;
      float v0 = s0[r] + sbias[d0];
      float v1 = s1[r] + sbias[d1];
      float t = fmaxf(v0, v1);
#pragma unroll
      for (int msk = 8; msk >= 1; msk >>= 1) t = fmaxf(t, __shfl_xor(t, msk, 32));
      float mnew = fmaxf(rmax[r], t);
      float alpha = __expf(rmax[r] - mnew);
      float p0 = __expf(v0 - mnew);
      float p1 = __expf(v1 - mnew);
      float ps = p0 + p1;
#pragma unroll
      for (int msk = 8; msk >= 1; msk >>= 1) ps += __shfl_xor(ps, msk, 32);
      rsum[r] = rsum[r] * alpha + ps;
      rmax[r] = mnew;
      o[0][r] *= alpha; o[1][r] *= alpha; o[2][r] *= alpha; o[3][r] *= alpha;
      // stash probabilities for C->A layout transpose (per-wave LDS region)
      pbuf[wave][r + 8 * hf][l16]      = (__bf16)p0;
      pbuf[wave][r + 8 * hf][16 + l16] = (__bf16)p1;
    }
    __builtin_amdgcn_wave_barrier(); // keep DS store->load order (same-wave LDS is in-order)

    // ---- P (16x32) as A fragment
    v16bf pf;
    {
      const __bf16* pp = &pbuf[wave][l16][16 * hf];
      pf = *(const v16bf*)pp;
    }

    // ---- O += P V : B col n = dim, Vt rows are key-contiguous
#pragma unroll
    for (int t = 0; t < 4; ++t) {
      v16bf vb = *(const v16bf*)(Vb + (size_t)(t * 16 + l16) * SEQ + kt + 16 * hf);
      o[t] = wmma_bf16(pf, vb, o[t]);
    }
  }

  // ---- normalize and write ctx[b, l, h*64+d]
#pragma unroll
  for (int t = 0; t < 4; ++t)
#pragma unroll
    for (int r = 0; r < 8; ++r) {
      int qrow = qBase + r + 8 * hf;
      int d = t * 16 + l16;
      float val = o[t][r] / rsum[r];
      ctx[((size_t)(b * SEQ + qrow)) * EMBED + h * HEAD_DIM + d] = (__bf16)val;
    }
}

// ---------------- host-side launch ----------------
extern "C" void kernel_launch(void* const* d_in, const int* in_sizes, int n_in,
                              void* d_out, int out_size, void* d_ws, size_t ws_size,
                              hipStream_t stream) {
  const float* x  = (const float*)d_in[0];
  const float* wq = (const float*)d_in[1];
  const float* bq = (const float*)d_in[2];
  const float* wk = (const float*)d_in[3];
  const float* bk = (const float*)d_in[4];
  const float* wv = (const float*)d_in[5];
  const float* bv = (const float*)d_in[6];
  const float* wo = (const float*)d_in[7];
  const float* bo = (const float*)d_in[8];
  const float* rel_bias = (const float*)d_in[9];
  float* out = (float*)d_out;

  const size_t NX = (size_t)MROWS * EMBED;
  const size_t NW = (size_t)EMBED * EMBED;
  const size_t NQKV = (size_t)BATCH * HEADS * SEQ * HEAD_DIM;

  char* p = (char*)d_ws;
  __bf16* xb  = (__bf16*)p; p += NX * 2;
  __bf16* wqb = (__bf16*)p; p += NW * 2;
  __bf16* wkb = (__bf16*)p; p += NW * 2;
  __bf16* wvb = (__bf16*)p; p += NW * 2;
  __bf16* wob = (__bf16*)p; p += NW * 2;
  __bf16* qb  = (__bf16*)p; p += NQKV * 2;
  __bf16* kb  = (__bf16*)p; p += NQKV * 2;
  __bf16* vtb = (__bf16*)p; p += NQKV * 2;
  __bf16* ctx = (__bf16*)p; p += NX * 2;
  float* bias_full = (float*)p; p += (size_t)HEADS * SEQ * 4;

  cvt_kernel<<<(int)((NX + 255) / 256), 256, 0, stream>>>(x, xb, (int)NX);
  cvt_kernel<<<(int)((NW + 255) / 256), 256, 0, stream>>>(wq, wqb, (int)NW);
  cvt_kernel<<<(int)((NW + 255) / 256), 256, 0, stream>>>(wk, wkb, (int)NW);
  cvt_kernel<<<(int)((NW + 255) / 256), 256, 0, stream>>>(wv, wvb, (int)NW);
  cvt_kernel<<<(int)((NW + 255) / 256), 256, 0, stream>>>(wo, wob, (int)NW);
  bias_kernel<<<(HEADS * SEQ + 255) / 256, 256, 0, stream>>>(rel_bias, bias_full);

  dim3 gg(EMBED / 128, MROWS / 128); // 8 x 64 blocks of 128x128
  proj_kernel<<<gg, 256, 0, stream>>>(xb, wqb, bq, qb, 0);
  proj_kernel<<<gg, 256, 0, stream>>>(xb, wkb, bk, kb, 1);
  proj_kernel<<<gg, 256, 0, stream>>>(xb, wvb, bv, vtb, 2);

  attn_kernel<<<dim3(SEQ / 128, HEADS, BATCH), 256, 0, stream>>>(qb, kb, vtb, bias_full, ctx);

  outproj_kernel<<<gg, 256, 0, stream>>>(ctx, wob, bo, out);
}
